// MultiheadSelfAttention_12953621365445
// MI455X (gfx1250) — matrix operand order repaired and checked by reading, not compiled, not run on hardware
//
#include <hip/hip_runtime.h>

// ---------------------------------------------------------------------------
// MI455X / gfx1250 fused multi-head self-attention.
//  - all matmuls on v_wmma_f32_16x16x32_f16 (fp32 accumulate, fp32 softmax)
//  - tiles staged LDS via global_load_async_to_lds_b128 (ASYNCcnt), double
//    buffered; fragments read from LDS with ds_load_b128
// ---------------------------------------------------------------------------

typedef __attribute__((ext_vector_type(16))) _Float16 v16h;
typedef __attribute__((ext_vector_type(8)))  _Float16 v8h;
typedef __attribute__((ext_vector_type(8)))  float    v8f;

namespace {
constexpr int kDim   = 1024;
constexpr int kHeads = 16;
constexpr int kHd    = 64;
constexpr int kSeq   = 2048;
constexpr int kBatch = 4;
constexpr int kTok   = kBatch * kSeq;      // 8192 tokens
constexpr int kQKVN  = 3 * kDim;           // 3072
}

// Low 32 bits of a generic pointer to LDS == byte offset from the group's
// LDS base (aperture truncation rule, ISA 10.2).
static __device__ __forceinline__ unsigned lds_off(const void* p) {
  return (unsigned)(unsigned long long)p;
}

// One wave-level async DMA: 32 lanes x 16B -> LDS.  GVS mode: uniform 64-bit
// SGPR base + per-lane 32-bit byte offset; vdst = per-lane LDS byte address.
static __device__ __forceinline__ void async_b128(unsigned lds_dst, unsigned voff,
                                                  unsigned long long sbase) {
  asm volatile("global_load_async_to_lds_b128 %0, %1, %2"
               :: "v"(lds_dst), "v"(voff), "s"(sbase)
               : "memory");
}

// ---- WMMA fragment helpers (layouts per CDNA5 ISA 7.12.2) ------------------

// 16-bit A-matrix 16x32: lane l -> row (l&15); halves K = kb..kb+7 and
// kb+16..kb+23, kb = (l>=16 ? 8 : 0).
__device__ __forceinline__ v16h frag_a_g(const _Float16* __restrict__ base,
                                         int ld, int m0, int k0, int lane) {
  const int r  = lane & 15;
  const int kb = (lane >> 4) << 3;
  const _Float16* p = base + (size_t)(m0 + r) * (size_t)ld + (size_t)(k0 + kb);
  v8h lo = *(const v8h*)(p);
  v8h hi = *(const v8h*)(p + 16);
  return __builtin_shufflevector(lo, hi, 0,1,2,3,4,5,6,7,8,9,10,11,12,13,14,15);
}

// A fragment from an LDS row pointer (32-wide k tile, row preselected).
__device__ __forceinline__ v16h frag_a_lds(const _Float16* rp, int lane) {
  const int kb = (lane >> 4) << 3;
  v8h lo = *(const v8h*)(rp + kb);
  v8h hi = *(const v8h*)(rp + kb + 16);
  return __builtin_shufflevector(lo, hi, 0,1,2,3,4,5,6,7,8,9,10,11,12,13,14,15);
}

// 16-bit B-matrix 32x16: lane l -> column (l&15); halves K = kb..kb+7 and
// kb+16..kb+23, kb = k0 + (l>=16 ? 8 : 0).  rp points at the Bt row (column) start.
__device__ __forceinline__ v16h frag_b_lds(const _Float16* rp, int k0, int lane) {
  const int kb = k0 + ((lane >> 4) << 3);
  v8h lo = *(const v8h*)(rp + kb);
  v8h hi = *(const v8h*)(rp + kb + 16);
  return __builtin_shufflevector(lo, hi, 0,1,2,3,4,5,6,7,8,9,10,11,12,13,14,15);
}

__device__ __forceinline__ v8f wmma_f16(v16h a, v16h b, v8f c) {
  return __builtin_amdgcn_wmma_f32_16x16x32_f16(false, a, false, b,
                                                (short)0, c, false, false);
}

__device__ __forceinline__ float rowmax16(float t) {
  t = fmaxf(t, __shfl_xor(t, 1));
  t = fmaxf(t, __shfl_xor(t, 2));
  t = fmaxf(t, __shfl_xor(t, 4));
  t = fmaxf(t, __shfl_xor(t, 8));
  return t;
}
__device__ __forceinline__ float rowsum16(float t) {
  t += __shfl_xor(t, 1);
  t += __shfl_xor(t, 2);
  t += __shfl_xor(t, 4);
  t += __shfl_xor(t, 8);
  return t;
}

// ---- prologue: casts / transposes -----------------------------------------

__global__ __launch_bounds__(256) void k_cast_f16(const float* __restrict__ src,
                                                  _Float16* __restrict__ dst, int n) {
  int i = blockIdx.x * 256 + threadIdx.x;
  if (i < n) dst[i] = (_Float16)src[i];
}

__global__ __launch_bounds__(256) void k_transpose_f16(const float* __restrict__ src,
                                                       _Float16* __restrict__ dst,
                                                       int rows, int cols) {
  int i = blockIdx.x * 256 + threadIdx.x;
  if (i < rows * cols) {
    int r = i / cols, c = i - r * cols;
    dst[(size_t)c * rows + r] = (_Float16)src[i];
  }
}

// ---- 128x128-tile GEMM core with async-LDS double buffering ----------------
// 8 waves/block; wave = 64x32 (4x2 WMMA tiles).  Waves 0..3 stage the A tile,
// waves 4..7 the B tile (4 async_b128 per wave per k-step).  LDS rows padded
// to 40 halves (bank-conflict-free 16B reads).

__device__ __forceinline__ void gemm_core_128x128(
    const _Float16* __restrict__ A,   // [M][K] f16 row-major
    const _Float16* __restrict__ Bt,  // [N][K] f16 row-major
    int K, int m0blk, int n0blk, v8f acc[4][2]) {
  __shared__ _Float16 sm[2][256 * 40];   // [buf][A rows 0..127 | B rows 128..255]

  const int tid  = threadIdx.x;
  const int lane = tid & 31;
  const int w    = tid >> 5;
  const int am   = (w >> 2) * 64;        // wave M offset in block tile
  const int bn   = (w & 3) * 32;         // wave N offset in block tile

  // staging role (wave-uniform): waves 0..3 -> A, waves 4..7 -> B
  const int  srow = tid & 127;
  const bool isB  = (tid >= 128);
  const unsigned voff = (unsigned)srow * (unsigned)(K * 2);   // row byte offset
  const _Float16* gbase = isB ? (Bt + (size_t)n0blk * K)
                              : (A  + (size_t)m0blk * K);
  const unsigned lrow0 = lds_off(&sm[0][((isB ? 128 : 0) + srow) * 40]);
  const unsigned lrow1 = lds_off(&sm[1][((isB ? 128 : 0) + srow) * 40]);

  auto issue = [&](int k0, unsigned lrow) {
    const unsigned long long sp = (unsigned long long)(gbase + k0);
#pragma unroll
    for (int t = 0; t < 4; ++t)
      async_b128(lrow + 16u * t, voff + 16u * t, sp);
  };

#pragma unroll
  for (int i = 0; i < 4; ++i)
#pragma unroll
    for (int j = 0; j < 2; ++j) acc[i][j] = (v8f){};

  issue(0, lrow0);
  for (int k0 = 0; k0 < K; k0 += 32) {
    const int buf = (k0 >> 5) & 1;
    if (k0 + 32 < K) {
      issue(k0 + 32, buf ? lrow0 : lrow1);
      asm volatile("s_wait_asynccnt 0x4" ::: "memory");   // current tile done
    } else {
      asm volatile("s_wait_asynccnt 0x0" ::: "memory");
    }
    __syncthreads();

    const _Float16* base = &sm[buf][0];
    v16h a[4], b[2];
#pragma unroll
    for (int i = 0; i < 4; ++i)
      a[i] = frag_a_lds(base + (am + 16 * i + (lane & 15)) * 40, lane);
#pragma unroll
    for (int j = 0; j < 2; ++j)
      b[j] = frag_b_lds(base + (128 + bn + 16 * j + (lane & 15)) * 40, 0, lane);
#pragma unroll
    for (int i = 0; i < 4; ++i)
#pragma unroll
      for (int j = 0; j < 2; ++j) acc[i][j] = wmma_f16(a[i], b[j], acc[i][j]);

    __syncthreads();   // everyone done reading before buf is overwritten
  }
}

// ---- QKV projection: [8192x1024] x [1024x3072] + bias ----------------------
// Epilogue scatters Q,K as [b,h,n,hd] f16 and V transposed as [b,h,hd,n] f16.

__global__ __launch_bounds__(256) void k_qkv_gemm(
    const _Float16* __restrict__ A, const _Float16* __restrict__ Bt,
    const float* __restrict__ bias,
    _Float16* __restrict__ Qb, _Float16* __restrict__ Kb,
    _Float16* __restrict__ Vt) {
  const int lane = threadIdx.x & 31;
  const int w    = threadIdx.x >> 5;
  const int m0blk = blockIdx.y * 128;
  const int n0blk = blockIdx.x * 128;

  v8f acc[4][2];
  gemm_core_128x128(A, Bt, kDim, m0blk, n0blk, acc);

  const int hl = lane >> 4, nl = lane & 15;
  const int m0 = m0blk + (w >> 2) * 64;
  const int n0 = n0blk + (w & 3) * 32;
#pragma unroll
  for (int i = 0; i < 4; ++i)
#pragma unroll
    for (int j = 0; j < 2; ++j)
#pragma unroll
      for (int r = 0; r < 8; ++r) {
        const int m   = m0 + 16 * i + r + 8 * hl;   // token
        const int col = n0 + 16 * j + nl;           // 0..3071
        const _Float16 hv = (_Float16)(acc[i][j][r] + bias[col]);
        const int which = col >> 10;                // 0=Q 1=K 2=V
        const int d  = col & (kDim - 1);
        const int h  = d >> 6;
        const int dd = d & (kHd - 1);
        const int bb = m >> 11;
        const int n  = m & (kSeq - 1);
        const size_t bh = (size_t)(bb * kHeads + h);
        if (which == 0)      Qb[(bh * kSeq + n) * kHd + dd] = hv;
        else if (which == 1) Kb[(bh * kSeq + n) * kHd + dd] = hv;
        else                 Vt[(bh * kHd + dd) * kSeq + n] = hv;
      }
}

// ---- flash attention -------------------------------------------------------
// Block = one (b,h) x 128 queries, 8 waves x 16 queries.  K/V key-blocks of 32
// staged in LDS by async DMA (waves 0..3 -> K tile, 4..7 -> V tile; 2 ops/wave),
// double buffered.  All 8 waves then share the LDS tiles (was 8x redundant
// global traffic).  Online softmax in fp32; P goes through a wave-private
// padded LDS tile to switch from C-layout to A-layout.

__global__ __launch_bounds__(256) void k_attn(
    const _Float16* __restrict__ Q,   // [b,h,n,hd]
    const _Float16* __restrict__ K,   // [b,h,n,hd]
    const _Float16* __restrict__ Vt,  // [b,h,hd,n]
    _Float16* __restrict__ Oh) {      // [b,n,1024] f16
  __shared__ _Float16 kt[2][32 * 72];   // K tile: 32 keys x 64 d, stride 72
  __shared__ _Float16 vtile[2][64 * 40];// V tile: 64 d x 32 keys, stride 40
  __shared__ _Float16 pt[8][16 * 40];   // per-wave P tile, stride 40

  const int tid  = threadIdx.x;
  const int lane = tid & 31;
  const int w    = tid >> 5;
  const int bh   = blockIdx.y;
  const int bb   = bh >> 4, h = bh & 15;
  const int q0   = blockIdx.x * 128 + w * 16;

  const _Float16* Qh = Q  + (size_t)bh * kSeq * kHd;
  const _Float16* Kh = K  + (size_t)bh * kSeq * kHd;
  const _Float16* Vh = Vt + (size_t)bh * kHd * kSeq;
  _Float16* pl = &pt[w][0];

  // staging descriptors (wave-uniform role; 2 chunks per thread)
  const bool isV = (tid >= 128);
  unsigned voffs[2], lrow0[2], lrow1[2];
#pragma unroll
  for (int t = 0; t < 2; ++t) {
    const int c = (tid & 127) * 2 + t;
    if (isV) {                       // V: 64 rows (d) x 4 x16B segs
      const int row = c >> 2, seg = c & 3;
      voffs[t] = (unsigned)row * 4096u + (unsigned)seg * 16u;  // Vt row = 4096B
      lrow0[t] = lds_off(&vtile[0][row * 40]) + seg * 16;
      lrow1[t] = lds_off(&vtile[1][row * 40]) + seg * 16;
    } else {                         // K: 32 rows (key) x 8 x16B segs
      const int row = c >> 3, seg = c & 7;
      voffs[t] = (unsigned)row * 128u + (unsigned)seg * 16u;   // K row = 128B
      lrow0[t] = lds_off(&kt[0][row * 72]) + seg * 16;
      lrow1[t] = lds_off(&kt[1][row * 72]) + seg * 16;
    }
  }
  auto issue = [&](int kb, int buf) {
    const unsigned long long sp = (unsigned long long)(
        isV ? (Vh + kb) : (Kh + (size_t)kb * kHd));
#pragma unroll
    for (int t = 0; t < 2; ++t)
      async_b128(buf ? lrow1[t] : lrow0[t], voffs[t], sp);
  };

  v16h qf[2];
  qf[0] = frag_a_g(Qh, kHd, q0, 0, lane);
  qf[1] = frag_a_g(Qh, kHd, q0, 32, lane);

  v8f o[4];
#pragma unroll
  for (int n = 0; n < 4; ++n) o[n] = (v8f){};
  float mrow[8], lrow[8];
#pragma unroll
  for (int r = 0; r < 8; ++r) { mrow[r] = -1e30f; lrow[r] = 0.f; }

  const int hl = lane >> 4, nl = lane & 15;

  issue(0, 0);
  for (int kb = 0; kb < kSeq; kb += 32) {
    const int buf = (kb >> 5) & 1;
    if (kb + 32 < kSeq) {
      issue(kb + 32, buf ^ 1);
      asm volatile("s_wait_asynccnt 0x2" ::: "memory");
    } else {
      asm volatile("s_wait_asynccnt 0x0" ::: "memory");
    }
    __syncthreads();

    // S = (Q K^T) / sqrt(hd); two 16-key subtiles from the shared K tile
    v8f s[2];
#pragma unroll
    for (int s2 = 0; s2 < 2; ++s2) {
      s[s2] = (v8f){};
      const _Float16* krp = &kt[buf][(s2 * 16 + nl) * 72];
      s[s2] = wmma_f16(qf[0], frag_b_lds(krp, 0, lane), s[s2]);
      s[s2] = wmma_f16(qf[1], frag_b_lds(krp, 32, lane), s[s2]);
    }
#pragma unroll
    for (int r = 0; r < 8; ++r) { s[0][r] *= 0.125f; s[1][r] *= 0.125f; }

    // online softmax (row = (r, lane-half); 16 lanes per row)
    float alpha[8];
#pragma unroll
    for (int r = 0; r < 8; ++r) {
      float t = rowmax16(fmaxf(s[0][r], s[1][r]));
      float mnew = fmaxf(mrow[r], t);
      alpha[r] = __expf(mrow[r] - mnew);
      mrow[r] = mnew;
      s[0][r] = __expf(s[0][r] - mnew);
      s[1][r] = __expf(s[1][r] - mnew);
      lrow[r] = lrow[r] * alpha[r] + rowsum16(s[0][r] + s[1][r]);
    }
#pragma unroll
    for (int n = 0; n < 4; ++n)
#pragma unroll
      for (int r = 0; r < 8; ++r) o[n][r] *= alpha[r];

    // C-layout -> A-layout via wave-private LDS tile
#pragma unroll
    for (int r = 0; r < 8; ++r) {
      pl[(r + 8 * hl) * 40 + nl]      = (_Float16)s[0][r];
      pl[(r + 8 * hl) * 40 + 16 + nl] = (_Float16)s[1][r];
    }
    asm volatile("s_wait_dscnt 0" ::: "memory");
    const _Float16* pr = pl + nl * 40 + hl * 8;
    v8h plo = *(const v8h*)(pr);
    v8h phi = *(const v8h*)(pr + 16);
    v16h pf = __builtin_shufflevector(plo, phi,
                                      0,1,2,3,4,5,6,7,8,9,10,11,12,13,14,15);

    // O += P @ V from the shared (pre-transposed) V tile
#pragma unroll
    for (int n = 0; n < 4; ++n)
      o[n] = wmma_f16(pf, frag_b_lds(&vtile[buf][(n * 16 + nl) * 40], 0, lane),
                      o[n]);

    __syncthreads();   // all waves done with buf before it is overwritten
  }

  // normalize, emit f16 row-major [b, n, 1024] for the output GEMM
#pragma unroll
  for (int r = 0; r < 8; ++r) {
    const float inv = 1.0f / lrow[r];
    const int q = q0 + r + 8 * hl;
    _Float16* dst = Oh + ((size_t)(bb * kSeq + q)) * kDim + h * kHd;
#pragma unroll
    for (int n = 0; n < 4; ++n)
      dst[n * 16 + nl] = (_Float16)(o[n][r] * inv);
  }
}

// ---- output projection: [8192x1024] x [1024x1024] + bias -> f32 ------------

__global__ __launch_bounds__(256) void k_proj_gemm(
    const _Float16* __restrict__ A, const _Float16* __restrict__ Bt,
    const float* __restrict__ bias, float* __restrict__ out) {
  const int lane = threadIdx.x & 31;
  const int w    = threadIdx.x >> 5;
  const int m0blk = blockIdx.y * 128;
  const int n0blk = blockIdx.x * 128;

  v8f acc[4][2];
  gemm_core_128x128(A, Bt, kDim, m0blk, n0blk, acc);

  const int hl = lane >> 4, nl = lane & 15;
  const int m0 = m0blk + (w >> 2) * 64;
  const int n0 = n0blk + (w & 3) * 32;
#pragma unroll
  for (int i = 0; i < 4; ++i)
#pragma unroll
    for (int j = 0; j < 2; ++j)
#pragma unroll
      for (int r = 0; r < 8; ++r) {
        const int m   = m0 + 16 * i + r + 8 * hl;
        const int col = n0 + 16 * j + nl;
        out[(size_t)m * kDim + col] = acc[i][j][r] + bias[col];
      }
}

// ---- host-side launcher ----------------------------------------------------

extern "C" void kernel_launch(void* const* d_in, const int* in_sizes, int n_in,
                              void* d_out, int out_size, void* d_ws, size_t ws_size,
                              hipStream_t stream) {
  const float* x      = (const float*)d_in[0];   // [4,2048,1024]
  const float* w_qkv  = (const float*)d_in[1];   // [1024,3072]
  const float* b_qkv  = (const float*)d_in[2];   // [3072]
  const float* w_proj = (const float*)d_in[3];   // [1024,1024]
  const float* b_proj = (const float*)d_in[4];   // [1024]
  float* out = (float*)d_out;

  char* ws = (char*)d_ws;
  const size_t MB = 1u << 20;
  _Float16* xh     = (_Float16*)(ws + 0);          // 16 MB
  _Float16* wqkvT  = (_Float16*)(ws + 16 * MB);    //  6 MB
  _Float16* wprojT = (_Float16*)(ws + 22 * MB);    //  2 MB
  _Float16* Qb     = (_Float16*)(ws + 24 * MB);    // 16 MB
  _Float16* Kb     = (_Float16*)(ws + 40 * MB);    // 16 MB
  _Float16* Vt     = (_Float16*)(ws + 56 * MB);    // 16 MB
  _Float16* Oh     = (_Float16*)(ws + 72 * MB);    // 16 MB  (total 88 MB)

  k_cast_f16<<<(kTok * kDim) / 256, 256, 0, stream>>>(x, xh, kTok * kDim);
  k_transpose_f16<<<(kDim * kQKVN) / 256, 256, 0, stream>>>(w_qkv, wqkvT, kDim, kQKVN);
  k_transpose_f16<<<(kDim * kDim) / 256, 256, 0, stream>>>(w_proj, wprojT, kDim, kDim);

  k_qkv_gemm<<<dim3(kQKVN / 128, kTok / 128), 256, 0, stream>>>(
      xh, wqkvT, b_qkv, Qb, Kb, Vt);

  k_attn<<<dim3(kSeq / 128, kBatch * kHeads), 256, 0, stream>>>(Qb, Kb, Vt, Oh);

  k_proj_gemm<<<dim3(kDim / 128, kTok / 128), 256, 0, stream>>>(
      Oh, wprojT, b_proj, out);
}
